// MyMultiheadAttention_77678778515968
// MI455X (gfx1250) — matrix operand order repaired and checked by reading, not compiled, not run on hardware
//
#include <hip/hip_runtime.h>

// ---------------------------------------------------------------------------
// MI455X (gfx1250) multihead attention forward, bf16 WMMA path.
// v_wmma_f32_16x16x32_bf16 with 2x4 register blocking + double-buffered
// (unroll-by-2, role-swapped) software pipelining -- no inter-buffer movs.
// V-tile staging uses GLOBAL_LOAD_ASYNC_TO_LDS_B128 when the builtin exists.
// ---------------------------------------------------------------------------

typedef __attribute__((ext_vector_type(16))) __bf16 v16bf;
typedef __attribute__((ext_vector_type(8)))  __bf16 v8bf;
typedef __attribute__((ext_vector_type(8)))  float  v8f;
typedef __attribute__((ext_vector_type(4)))  int    v4i;

#define T_LEN 2048
#define S_LEN 2048
#define BATCH 4
#define EMB   1024
#define HEADS 16
#define HD    64
#define MROWS (T_LEN * BATCH)   // 8192 rows for the projection GEMMs

#if __has_builtin(__builtin_amdgcn_global_load_async_to_lds_b128)
#define HAVE_ASYNC_LDS 1
#else
#define HAVE_ASYNC_LDS 0
#endif

// ---- A-operand loader: 16-bit A matrix 16x32 per wave -----------------------
// lane L: row = L&15, g = L>>4; elements 0..7  <-> K = k0 + g*16 + 0..7
//                               elements 8..15 <-> K = k0 + g*16 + 8 + 0..7
__device__ __forceinline__ v16bf load_a16(const __bf16* __restrict__ rowptr,
                                          int k0, int g) {
  v8bf lo = *(const v8bf*)(rowptr + k0 + g * 16);
  v8bf hi = *(const v8bf*)(rowptr + k0 + g * 16 + 8);
  v16bf a;
#pragma unroll
  for (int i = 0; i < 8; ++i) { a[i] = lo[i]; a[8 + i] = hi[i]; }
  return a;
}

__device__ __forceinline__ v8f wmma_bf16(v16bf a, v16bf b, v8f c) {
  return __builtin_amdgcn_wmma_f32_16x16x32_bf16(false, a, false, b,
                                                 (short)0, c, false, false);
}

// ---------------------------------------------------------------------------
__global__ void zero_f32_kernel(float* __restrict__ p, size_t n) {
  size_t i = (size_t)blockIdx.x * blockDim.x + threadIdx.x;
  size_t stride = (size_t)gridDim.x * blockDim.x;
  for (; i < n; i += stride) p[i] = 0.0f;
}

__global__ void cast_f32_bf16_kernel(const float* __restrict__ src,
                                     __bf16* __restrict__ dst, size_t n) {
  size_t i = (size_t)blockIdx.x * blockDim.x + threadIdx.x;
  size_t stride = (size_t)gridDim.x * blockDim.x;
  for (; i < n; i += stride) dst[i] = (__bf16)src[i];
}

// ---------------------------------------------------------------------------
// Pipelined GEMM core: per-wave tile 32(M) x 64(N), K = EMB, bf16 operands,
// f32 accumulators. Double-buffered with role swap: buffer X covers k0 while
// buffer Y is loading k0+32, then roles flip -- zero register copies.
// ---------------------------------------------------------------------------
struct GemmAcc { v8f c[2][4]; };

__device__ __forceinline__ GemmAcc gemm_core(const __bf16* __restrict__ A,
                                             const __bf16* __restrict__ W,
                                             int mbase, int nbase,
                                             int g, int ln) {
  const __bf16* arow0 = A + (size_t)(mbase + ln) * EMB;
  const __bf16* arow1 = A + (size_t)(mbase + 16 + ln) * EMB;
  const __bf16* wrow  = W + (size_t)(nbase + ln) * EMB + g * 16;

  GemmAcc acc;
#pragma unroll
  for (int mt = 0; mt < 2; ++mt)
#pragma unroll
    for (int nt = 0; nt < 4; ++nt) acc.c[mt][nt] = (v8f)0.0f;

  // prologue: buffer X holds k = 0
  v16bf xa0 = load_a16(arow0, 0, g);
  v16bf xa1 = load_a16(arow1, 0, g);
  v16bf xb[4];
#pragma unroll
  for (int nt = 0; nt < 4; ++nt)
    xb[nt] = *(const v16bf*)(wrow + (size_t)nt * 16 * EMB);

  v16bf ya0, ya1, yb[4];
  int k0 = 0;
  for (; k0 + 64 < EMB; k0 += 64) {
    // load Y @ k0+32, consume X @ k0
    ya0 = load_a16(arow0, k0 + 32, g);
    ya1 = load_a16(arow1, k0 + 32, g);
#pragma unroll
    for (int nt = 0; nt < 4; ++nt)
      yb[nt] = *(const v16bf*)(wrow + (size_t)nt * 16 * EMB + k0 + 32);
#pragma unroll
    for (int nt = 0; nt < 4; ++nt) {
      acc.c[0][nt] = wmma_bf16(xa0, xb[nt], acc.c[0][nt]);
      acc.c[1][nt] = wmma_bf16(xa1, xb[nt], acc.c[1][nt]);
    }
    // load X @ k0+64, consume Y @ k0+32
    xa0 = load_a16(arow0, k0 + 64, g);
    xa1 = load_a16(arow1, k0 + 64, g);
#pragma unroll
    for (int nt = 0; nt < 4; ++nt)
      xb[nt] = *(const v16bf*)(wrow + (size_t)nt * 16 * EMB + k0 + 64);
#pragma unroll
    for (int nt = 0; nt < 4; ++nt) {
      acc.c[0][nt] = wmma_bf16(ya0, yb[nt], acc.c[0][nt]);
      acc.c[1][nt] = wmma_bf16(ya1, yb[nt], acc.c[1][nt]);
    }
  }
  // tail: k0 == EMB-64; X holds k0, load Y @ k0+32, then drain both
  ya0 = load_a16(arow0, k0 + 32, g);
  ya1 = load_a16(arow1, k0 + 32, g);
#pragma unroll
  for (int nt = 0; nt < 4; ++nt)
    yb[nt] = *(const v16bf*)(wrow + (size_t)nt * 16 * EMB + k0 + 32);
#pragma unroll
  for (int nt = 0; nt < 4; ++nt) {
    acc.c[0][nt] = wmma_bf16(xa0, xb[nt], acc.c[0][nt]);
    acc.c[1][nt] = wmma_bf16(xa1, xb[nt], acc.c[1][nt]);
  }
#pragma unroll
  for (int nt = 0; nt < 4; ++nt) {
    acc.c[0][nt] = wmma_bf16(ya0, yb[nt], acc.c[0][nt]);
    acc.c[1][nt] = wmma_bf16(ya1, yb[nt], acc.c[1][nt]);
  }
  return acc;
}

// ---------------------------------------------------------------------------
// Projection GEMM: dst = (A[M,K] @ W[N,K]^T) * scale, bf16 out, head-major:
// dst[((b*H + h)*LEN + t)*64 + d], row m = t*BATCH + b, col f = h*64 + d.
// ---------------------------------------------------------------------------
__global__ __launch_bounds__(128)
void gemm_proj_kernel(const __bf16* __restrict__ A, const __bf16* __restrict__ W,
                      __bf16* __restrict__ dst, float scale) {
  const int lane = threadIdx.x & 31;
  const int wave = threadIdx.x >> 5;
  const int g    = lane >> 4;
  const int ln   = lane & 15;
  const int mbase = (blockIdx.x * 4 + wave) * 32;
  const int nbase = blockIdx.y * 64;

  GemmAcc acc = gemm_core(A, W, mbase, nbase, g, ln);

#pragma unroll
  for (int mt = 0; mt < 2; ++mt)
#pragma unroll
    for (int nt = 0; nt < 4; ++nt)
#pragma unroll
      for (int v = 0; v < 8; ++v) {
        int m = mbase + mt * 16 + v + 8 * g;
        int f = nbase + nt * 16 + ln;
        int t = m / BATCH, b = m % BATCH;
        int h = f / HD,    d = f % HD;
        dst[((size_t)(b * HEADS + h) * T_LEN + t) * HD + d] =
            (__bf16)(acc.c[mt][nt][v] * scale);
      }
}

// ---------------------------------------------------------------------------
// Output projection GEMM: out[m,f] = A[m,:] . Wo[f,:] + bias[f]  (f32 out)
// ---------------------------------------------------------------------------
__global__ __launch_bounds__(128)
void gemm_out_kernel(const __bf16* __restrict__ A, const __bf16* __restrict__ W,
                     const float* __restrict__ bias, float* __restrict__ out) {
  const int lane = threadIdx.x & 31;
  const int wave = threadIdx.x >> 5;
  const int g    = lane >> 4;
  const int ln   = lane & 15;
  const int mbase = (blockIdx.x * 4 + wave) * 32;
  const int nbase = blockIdx.y * 64;

  GemmAcc acc = gemm_core(A, W, mbase, nbase, g, ln);

#pragma unroll
  for (int mt = 0; mt < 2; ++mt)
#pragma unroll
    for (int nt = 0; nt < 4; ++nt)
#pragma unroll
      for (int v = 0; v < 8; ++v) {
        int m = mbase + mt * 16 + v + 8 * g;
        int f = nbase + nt * 16 + ln;
        out[(size_t)m * EMB + f] = acc.c[mt][nt][v] + bias[f];
      }
}

// ---------------------------------------------------------------------------
// Attention: one (b, h, 16-row t-tile) per wave. Two-pass online softmax.
// ---------------------------------------------------------------------------
__global__ __launch_bounds__(128)
void attn_kernel(const __bf16* __restrict__ Qh, const __bf16* __restrict__ Kh,
                 const __bf16* __restrict__ Vh, __bf16* __restrict__ ctx,
                 float* __restrict__ avg) {
  __shared__ __align__(16) __bf16 pt[4][16][32];   // P tiles, one per wave
  __shared__ __align__(16) __bf16 vt[4][32][72];   // V tiles (padded rows)

  const int lane = threadIdx.x & 31;
  const int w    = threadIdx.x >> 5;
  const int g    = lane >> 4;
  const int ln   = lane & 15;
  const int b    = blockIdx.z;
  const int h    = blockIdx.y;
  const int tbase = (blockIdx.x * 4 + w) * 16;

  const size_t hoff = (size_t)(b * HEADS + h) * T_LEN * HD;
  const __bf16* Qp = Qh + hoff;
  const __bf16* Kp = Kh + hoff;
  const __bf16* Vp = Vh + hoff;

  // Q A-operand for the full K=64 head dim (two 16x32 steps), kept in VGPRs.
  v16bf qa[2];
#pragma unroll
  for (int ks = 0; ks < 2; ++ks)
    qa[ks] = load_a16(Qp + (size_t)(tbase + ln) * HD, ks * 32, g);

  float mstat[8], lstat[8];
#pragma unroll
  for (int v = 0; v < 8; ++v) { mstat[v] = -3.0e38f; lstat[v] = 0.0f; }

  // ---- pass 1: softmax statistics (2 s-subtiles per step, loads hoisted) ----
  for (int s0 = 0; s0 < S_LEN; s0 += 32) {
    v16bf bk[2][2];
#pragma unroll
    for (int sub = 0; sub < 2; ++sub)
#pragma unroll
      for (int ks = 0; ks < 2; ++ks)
        bk[sub][ks] = *(const v16bf*)(Kp + (size_t)(s0 + sub * 16 + ln) * HD +
                                      ks * 32 + g * 16);
    v8f c[2];
#pragma unroll
    for (int sub = 0; sub < 2; ++sub) {
      c[sub] = (v8f)0.0f;
      c[sub] = wmma_bf16(qa[0], bk[sub][0], c[sub]);
      c[sub] = wmma_bf16(qa[1], bk[sub][1], c[sub]);
    }
#pragma unroll
    for (int sub = 0; sub < 2; ++sub)
#pragma unroll
      for (int v = 0; v < 8; ++v) {
        float tmax = c[sub][v];
#pragma unroll
        for (int off = 1; off < 16; off <<= 1)
          tmax = fmaxf(tmax, __shfl_xor(tmax, off, 16));
        float mnew = fmaxf(mstat[v], tmax);
        float e = __expf(c[sub][v] - mnew);
        float esum = e;
#pragma unroll
        for (int off = 1; off < 16; off <<= 1)
          esum += __shfl_xor(esum, off, 16);
        lstat[v] = lstat[v] * __expf(mstat[v] - mnew) + esum;
        mstat[v] = mnew;
      }
  }
#pragma unroll
  for (int v = 0; v < 8; ++v) lstat[v] = 1.0f / lstat[v];

  // avg_weights mapping faithful to reference: n = b*H + h contributes to
  // batch index n % B (reshape(H, B, T, S) of the [B*H, T, S] buffer).
  float* avgrow = avg + (size_t)((b * HEADS + h) % BATCH) * T_LEN * S_LEN;

  // ---- pass 2: probabilities, avg accumulation, P @ V ----
  v8f o[4];
#pragma unroll
  for (int nt = 0; nt < 4; ++nt) o[nt] = (v8f)0.0f;

  for (int s0 = 0; s0 < S_LEN; s0 += 32) {
    // stage V[s0..s0+31][0..63] into LDS (each lane owns one 128B row)
    {
      const __bf16* vrow = Vp + (size_t)(s0 + lane) * HD;
      __bf16* lrow = &vt[w][lane][0];
#if HAVE_ASYNC_LDS
#pragma unroll
      for (int i = 0; i < 8; ++i)
        __builtin_amdgcn_global_load_async_to_lds_b128(
            (v4i*)(vrow + i * 8), (v4i*)(lrow + i * 8), 0, 0);
#else
#pragma unroll
      for (int i = 0; i < 8; ++i)
        *(v8bf*)&lrow[i * 8] = *(const v8bf*)(vrow + i * 8);
#endif
    }
    // scores for both s-subtiles, loads hoisted ahead of the WMMA chain
    v16bf bk[2][2];
#pragma unroll
    for (int sub = 0; sub < 2; ++sub)
#pragma unroll
      for (int ks = 0; ks < 2; ++ks)
        bk[sub][ks] = *(const v16bf*)(Kp + (size_t)(s0 + sub * 16 + ln) * HD +
                                      ks * 32 + g * 16);
#pragma unroll
    for (int sub = 0; sub < 2; ++sub) {
      v8f c = (v8f)0.0f;
      c = wmma_bf16(qa[0], bk[sub][0], c);
      c = wmma_bf16(qa[1], bk[sub][1], c);
#pragma unroll
      for (int v = 0; v < 8; ++v) {
        float p = __expf(c[v] - mstat[v]) * lstat[v];
        int r = v + 8 * g;                        // C-tile row for this lane
        pt[w][r][sub * 16 + ln] = (__bf16)p;
        size_t aidx = (size_t)(tbase + r) * S_LEN + (s0 + sub * 16 + ln);
        __hip_atomic_fetch_add(avgrow + aidx, p * (1.0f / HEADS),
                               __ATOMIC_RELAXED, __HIP_MEMORY_SCOPE_AGENT);
      }
    }
#if HAVE_ASYNC_LDS
    asm volatile("s_wait_asynccnt 0x0" ::: "memory");
#endif
    __syncthreads();
    // A operand from staged P (16x32 A layout, contiguous 16B chunks)
    v16bf pa;
    {
      v8bf lo = *(const v8bf*)&pt[w][ln][g * 16];
      v8bf hi = *(const v8bf*)&pt[w][ln][g * 16 + 8];
#pragma unroll
      for (int i = 0; i < 8; ++i) { pa[i] = lo[i]; pa[8 + i] = hi[i]; }
    }
#pragma unroll
    for (int nt = 0; nt < 4; ++nt) {
      // B operand: B[k=s_local][n=d] = V[s0+k][nt*16+ln], gathered from LDS.
      v16bf bv;
      int d = nt * 16 + ln;
#pragma unroll
      for (int j = 0; j < 16; ++j) bv[j] = vt[w][g * 16 + j][d];
      o[nt] = wmma_bf16(pa, bv, o[nt]);
    }
    __syncthreads();
  }

  // write context bf16 in [t, b, e] layout for the out-projection GEMM
#pragma unroll
  for (int nt = 0; nt < 4; ++nt)
#pragma unroll
    for (int v = 0; v < 8; ++v) {
      int r = v + 8 * g;
      int e = h * HD + nt * 16 + ln;
      ctx[((size_t)(tbase + r) * BATCH + b) * EMB + e] = (__bf16)o[nt][v];
    }
}

// ---------------------------------------------------------------------------
extern "C" void kernel_launch(void* const* d_in, const int* in_sizes, int n_in,
                              void* d_out, int out_size, void* d_ws, size_t ws_size,
                              hipStream_t stream) {
  const float* q    = (const float*)d_in[0];
  const float* k    = (const float*)d_in[1];
  const float* v    = (const float*)d_in[2];
  const float* wq   = (const float*)d_in[3];
  const float* wk   = (const float*)d_in[4];
  const float* wv   = (const float*)d_in[5];
  const float* wo   = (const float*)d_in[6];
  const float* bias = (const float*)d_in[7];

  float* out = (float*)d_out;                       // [T,B,E] f32
  float* avg = out + (size_t)MROWS * EMB;           // [B,T,S] f32

  // workspace layout (bf16 elements): 4 weights + act buf + Qh/Kh/Vh + ctx
  __bf16* ws = (__bf16*)d_ws;
  const size_t WE = (size_t)EMB * EMB;              // 1M elems per weight
  const size_t AE = (size_t)MROWS * EMB;            // 8M elems
  const size_t HE = (size_t)BATCH * HEADS * T_LEN * HD;  // 8M elems
  __bf16* Wq = ws;
  __bf16* Wk = ws + WE;
  __bf16* Wv = ws + 2 * WE;
  __bf16* Wo = ws + 3 * WE;
  __bf16* Ab = ws + 4 * WE;        // reused cast buffer for q/k/v inputs
  __bf16* Qh = Ab + AE;
  __bf16* Kh = Qh + HE;
  __bf16* Vh = Kh + HE;
  __bf16* Cx = Vh + HE;

  dim3 blk(128);
  dim3 ggrid(MROWS / 128, EMB / 64);                // 32-row wave tiles x4
  dim3 agrid(T_LEN / 64, HEADS, BATCH);             // attention grid

  zero_f32_kernel<<<2048, 256, 0, stream>>>(avg, (size_t)BATCH * T_LEN * S_LEN);

  cast_f32_bf16_kernel<<<1024, 256, 0, stream>>>(wq, Wq, WE);
  cast_f32_bf16_kernel<<<1024, 256, 0, stream>>>(wk, Wk, WE);
  cast_f32_bf16_kernel<<<1024, 256, 0, stream>>>(wv, Wv, WE);
  cast_f32_bf16_kernel<<<1024, 256, 0, stream>>>(wo, Wo, WE);

  cast_f32_bf16_kernel<<<2048, 256, 0, stream>>>(q, Ab, AE);
  gemm_proj_kernel<<<ggrid, blk, 0, stream>>>(Ab, Wq, Qh, 0.125f);  // hd^-0.5

  cast_f32_bf16_kernel<<<2048, 256, 0, stream>>>(k, Ab, AE);
  gemm_proj_kernel<<<ggrid, blk, 0, stream>>>(Ab, Wk, Kh, 1.0f);

  cast_f32_bf16_kernel<<<2048, 256, 0, stream>>>(v, Ab, AE);
  gemm_proj_kernel<<<ggrid, blk, 0, stream>>>(Ab, Wv, Vh, 1.0f);

  attn_kernel<<<agrid, blk, 0, stream>>>(Qh, Kh, Vh, Cx, avg);

  gemm_out_kernel<<<ggrid, blk, 0, stream>>>(Cx, Wo, bias, out);
}
